// InteractionLayer_76897094467605
// MI455X (gfx1250) — matrix operand order repaired and checked by reading, not compiled, not run on hardware
//
#include <hip/hip_runtime.h>
#include <hip/hip_bf16.h>
#include <math.h>

typedef _Float16 v16h __attribute__((ext_vector_type(16)));
typedef _Float16 v8h  __attribute__((ext_vector_type(8)));
typedef float    v8f  __attribute__((ext_vector_type(8)));

#define KP 1032                       // padded K row length in f16 (1024 + 8 pad -> kills bank conflicts)
#define WT_DIR_STRIDE (2 * 32 * KP)   // f16 elems per direction: [cat][w][KP]
#define INV_SQRT3F 0.57735026918962576f

// ---------------------------------------------------------------------------
// Prep: build f16, transposed ([cat][w][k]) and pre-scaled weight tables in ws.
// cat0 rows: k<512 -> W0[u,v,w]*s ; k>=512 -> W1[u,v,w]*s/sqrt(3)   (out0 GEMM)
// cat1 rows: k<512 -> W2[u,v,w]*s ; k>=512 -> W3[u,v,w]*s           (out1 GEMM)
// ---------------------------------------------------------------------------
__global__ __launch_bounds__(256) void prep_weights(const float* __restrict__ Ws2d,
                                                    const float* __restrict__ Wd2s,
                                                    _Float16* __restrict__ wt,
                                                    float scale) {
  const int total = 2 * WT_DIR_STRIDE;
  for (int idx = blockIdx.x * blockDim.x + threadIdx.x; idx < total;
       idx += gridDim.x * blockDim.x) {
    int k   = idx % KP;
    int w   = (idx / KP) & 31;
    int cat = (idx / (KP * 32)) & 1;
    int dir = idx / (KP * 64);
    float f = 0.f;
    if (k < 1024) {
      int half = k >> 9;
      int u = (k >> 5) & 15;
      int v = k & 31;
      int path = cat ? (2 + half) : half;
      const float* W = dir ? Wd2s : Ws2d;
      f = W[((path * 16 + u) * 32 + v) * 32 + w] * scale;
      if (cat == 0 && half == 1) f *= INV_SQRT3F;
    }
    wt[idx] = (_Float16)f;
  }
}

// ---------------------------------------------------------------------------
// Init: d_out = concat(y0, y1); scatter kernels then atomically accumulate.
// ---------------------------------------------------------------------------
__global__ void init_out(const float4* __restrict__ y0, const float4* __restrict__ y1,
                         float4* __restrict__ out, int n0q, int n1q) {
  const int stride = gridDim.x * blockDim.x;
  const int t = blockIdx.x * blockDim.x + threadIdx.x;
  for (int i = t; i < n0q; i += stride) out[i] = y0[i];
  for (int i = t; i < n1q; i += stride) out[n0q + i] = y1[i];
}

__device__ __forceinline__ v16h loadB16(const _Float16* p) {
  v8h lo = *(const v8h*)(p);
  v8h hi = *(const v8h*)(p + 8);
  return __builtin_shufflevector(lo, hi, 0,1,2,3,4,5,6,7,8,9,10,11,12,13,14,15);
}

__device__ __forceinline__ void wmma2(const v16h& a, const _Float16* bA, const _Float16* bB,
                                      int k0, v8f& acc0, v8f& acc1) {
  v16h b0 = loadB16(bA + k0);
  v16h b1 = loadB16(bB + k0);
  acc0 = __builtin_amdgcn_wmma_f32_16x16x32_f16(false, a, false, b0, (short)0, acc0, false, false);
  acc1 = __builtin_amdgcn_wmma_f32_16x16x32_f16(false, a, false, b1, (short)0, acc1, false, false);
}

// ---------------------------------------------------------------------------
// Main fused kernel: 256 threads = 8 waves sharing ONE LDS weight copy.
// Waves 0-3 -> edge-tile slot 0, waves 4-7 -> edge-tile slot 1 (32 edges/iter).
// Within a slot: role 0 -> out0 (scalars), roles 1..3 -> out1 i=0,1,2.
// Each wave: GEMM M=16 edges, N=32, K=1024 via v_wmma_f32_16x16x32_f16.
// ---------------------------------------------------------------------------
__global__ __launch_bounds__(256) void fctp_main(const float* __restrict__ ef,
                                                 const float* __restrict__ ytab,
                                                 const int* __restrict__ gIdx,
                                                 const int* __restrict__ sIdx,
                                                 const _Float16* __restrict__ wtg,
                                                 float* __restrict__ outdst,
                                                 int E) {
  __shared__ _Float16 Wt[2][32][KP];    // 132,096 B: f16 weights, [cat][w][k]
  __shared__ float x1s[2][16][68];      // edge features per slot, padded rows
  __shared__ float x2s[2][16][132];     // gathered node features per slot, padded rows
  __shared__ int   ss[2][16];           // scatter node indices per slot

  const int tid  = threadIdx.x;
  const int lane = tid & 31;
  const int wave = tid >> 5;            // 0..7
  const int slot = wave >> 2;           // which 16-edge tile of the 32-edge pair
  const int role = wave & 3;            // 0: out0, 1..3: out1 i = role-1
  const int row  = lane & 15;           // edge row within tile (A/M index)
  const int khi  = lane >> 4;           // which K half-slice this lane holds
  const int n    = lane & 15;           // output column within N-tile (B/C index)

  // ---- one-time: copy this direction's weights into LDS (contiguous, wide) ----
  {
    const float4* s4 = (const float4*)wtg;
    float4* d4 = (float4*)&Wt[0][0][0];
    const int nq = WT_DIR_STRIDE / 8;   // 8 f16 per float4
    for (int i = tid; i < nq; i += 256) d4[i] = s4[i];
  }
  __syncthreads();

  // Per-lane A-fragment K mapping for v_wmma_f32_16x16x32_f16:
  // elements 0..15 -> K = 16*khi + j
  int Ka[16];
#pragma unroll
  for (int j = 0; j < 16; ++j) Ka[j] = 16 * khi + j;

  const int cat = (role == 0) ? 0 : 1;
  const _Float16* bA = &Wt[cat][n][0];
  const _Float16* bB = &Wt[cat][n + 16][0];
  const float (*x1r)[68]  = x1s[slot];
  const float (*x2r)[132] = x2s[slot];

  const int npairs = (E + 31) >> 5;     // 32 edges per block iteration
  for (int pair = blockIdx.x; pair < npairs; pair += gridDim.x) {
    const long e0 = (long)pair << 5;

    // ---- stage 32 edges: x1 (64 f32) + gathered x2 (128 f32), 8 threads/edge ----
    {
      const int er = tid >> 3;          // 0..31
      const int sl = er >> 4;
      const int rw = er & 15;
      const int q  = tid & 7;
      const long e = e0 + er;
      float4 z; z.x = z.y = z.z = z.w = 0.f;
      float4 a0 = z, a1 = z, c0 = z, c1 = z, c2 = z, c3 = z;
      if (e < (long)E) {
        const float4* p1 = (const float4*)(ef + e * 64) + q * 2;
        a0 = p1[0]; a1 = p1[1];
        const int g = gIdx[e];
        const float4* p2 = (const float4*)(ytab + (long)g * 128) + q * 4;
        c0 = p2[0]; c1 = p2[1]; c2 = p2[2]; c3 = p2[3];
      }
      float4* d1 = (float4*)&x1s[sl][rw][q * 8];
      d1[0] = a0; d1[1] = a1;
      float4* d2 = (float4*)&x2s[sl][rw][q * 16];
      d2[0] = c0; d2[1] = c1; d2[2] = c2; d2[3] = c3;
      if (tid < 32) {
        const long e2 = e0 + tid;
        ss[tid >> 4][tid & 15] = (e2 < (long)E) ? sIdx[e2] : 0;
      }
    }
    __syncthreads();

    v8f acc0 = {}, acc1 = {};

    if (role == 0) {
      // ---- out0: K half 0 = x1_0[u]*x2_0[v]; half 1 = sum_i x1_1[u,i]*x2_1[v,i]
      float xr[16], r0[16], r1[16], r2[16];
#pragma unroll
      for (int j = 0; j < 16; ++j) {
        const int v = Ka[j];
        xr[j] = x2r[row][v];
        r0[j] = x2r[row][32 + 3 * v + 0];
        r1[j] = x2r[row][32 + 3 * v + 1];
        r2[j] = x2r[row][32 + 3 * v + 2];
      }
      for (int u = 0; u < 16; ++u) {
        const float x1u = x1r[row][u];
        v16h a;
#pragma unroll
        for (int j = 0; j < 16; ++j) a[j] = (_Float16)(x1u * xr[j]);
        wmma2(a, bA, bB, u * 32 + 16 * khi, acc0, acc1);
      }
      for (int u = 0; u < 16; ++u) {
        const float xa = x1r[row][16 + 3 * u + 0];
        const float xb = x1r[row][16 + 3 * u + 1];
        const float xc = x1r[row][16 + 3 * u + 2];
        v16h a;
#pragma unroll
        for (int j = 0; j < 16; ++j)
          a[j] = (_Float16)(xa * r0[j] + xb * r1[j] + xc * r2[j]);
        wmma2(a, bA, bB, 512 + u * 32 + 16 * khi, acc0, acc1);
      }
    } else {
      // ---- out1 component iv: half 0 = x1_0[u]*x2_1[v,iv]; half 1 = x1_1[u,iv]*x2_0[v]
      const int iv = role - 1;
      float xrA[16], xrB[16];
#pragma unroll
      for (int j = 0; j < 16; ++j) {
        const int v = Ka[j];
        xrA[j] = x2r[row][32 + 3 * v + iv];
        xrB[j] = x2r[row][v];
      }
      for (int u = 0; u < 16; ++u) {
        const float x1u = x1r[row][u];
        v16h a;
#pragma unroll
        for (int j = 0; j < 16; ++j) a[j] = (_Float16)(x1u * xrA[j]);
        wmma2(a, bA, bB, u * 32 + 16 * khi, acc0, acc1);
      }
      for (int u = 0; u < 16; ++u) {
        const float x1ui = x1r[row][16 + 3 * u + iv];
        v16h a;
#pragma unroll
        for (int j = 0; j < 16; ++j) a[j] = (_Float16)(x1ui * xrB[j]);
        wmma2(a, bA, bB, 512 + u * 32 + 16 * khi, acc0, acc1);
      }
    }

    // ---- SiLU + scatter-add (C layout: lane col n; VGPR j -> edge row j + 8*khi)
#pragma unroll
    for (int t = 0; t < 2; ++t) {
      const v8f& c = t ? acc1 : acc0;
      const int wcol = n + 16 * t;
      const int ch = (role == 0) ? wcol : (32 + 3 * wcol + (role - 1));
#pragma unroll
      for (int j = 0; j < 8; ++j) {
        const int m = j + 8 * khi;
        const float x = c[j];
        const float y = x / (1.f + __expf(-x));
        atomicAdd(outdst + (long)ss[slot][m] * 128 + ch, y);
      }
    }
    __syncthreads();
  }
}

// ---------------------------------------------------------------------------
extern "C" void kernel_launch(void* const* d_in, const int* in_sizes, int n_in,
                              void* d_out, int out_size, void* d_ws, size_t ws_size,
                              hipStream_t stream) {
  const float* ef   = (const float*)d_in[0];
  const float* y0   = (const float*)d_in[1];
  const float* y1   = (const float*)d_in[2];
  const float* Ws2d = (const float*)d_in[3];
  const float* Wd2s = (const float*)d_in[4];
  const int*   src  = (const int*)d_in[5];
  const int*   dst  = (const int*)d_in[6];

  const int E  = in_sizes[0] / 64;
  const int N0 = in_sizes[1] / 128;
  const int N1 = in_sizes[2] / 128;

  // Fold INV_SQRT_FAN (=1/32) and 1/sqrt(N0+N1) into the weights.
  const float scale = (1.0f / 32.0f) / sqrtf((float)(N0 + N1));

  _Float16* wt = (_Float16*)d_ws;

  prep_weights<<<128, 256, 0, stream>>>(Ws2d, Wd2s, wt, scale);
  init_out<<<512, 256, 0, stream>>>((const float4*)y0, (const float4*)y1,
                                    (float4*)d_out, N0 * 32, N1 * 32);

  float* out = (float*)d_out;
  // s2d: gather y0[src], scatter by dst into the (y1 + s2d) half
  fctp_main<<<2048, 256, 0, stream>>>(ef, y0, src, dst, wt, out + (size_t)N0 * 128, E);
  // d2s: gather y1[dst], scatter by src into the (y0 + d2s) half
  fctp_main<<<2048, 256, 0, stream>>>(ef, y1, dst, src, wt + WT_DIR_STRIDE, out, E);
}
